// PointConv_21715354650018
// MI455X (gfx1250) — matrix operand order repaired and checked by reading, not compiled, not run on hardware
//
#include <hip/hip_runtime.h>
#include <math.h>

// ---------------- problem constants (match reference) ----------------
#define BB    4
#define CC    64
#define NN    4096
#define KK    16
#define WW    32      // WeightNet out channels
#define COUT  64
#define DN    16      // DensityNet hidden channels
#define EPSF  1e-5f
#define TP    16      // points per tile in fused kernel
#define BN    (BB*NN) // 16384 points total
#define LINSZ (COUT*CC*WW) // 64*2048

typedef __attribute__((ext_vector_type(16))) _Float16 v16h;
typedef __attribute__((ext_vector_type(8)))  _Float16 v8h;
typedef __attribute__((ext_vector_type(8)))  float    v8f;

__device__ __forceinline__ v8f wmma_f16(const v16h a, const v16h b, const v8f c) {
  // D = A(16x32 f16) x B(32x16 f16) + C(16x16 f32)
  return __builtin_amdgcn_wmma_f32_16x16x32_f16(false, a, false, b, (short)0, c, false, false);
}

// ---------------- 1) squared norms of positions ----------------
__global__ void sqnorm_kernel(const float* __restrict__ pos, float* __restrict__ sq) {
  int pid = blockIdx.x * blockDim.x + threadIdx.x;
  if (pid >= BN) return;
  int b = pid / NN, n = pid % NN;
  const float* pb = pos + (size_t)b * 3 * NN;
  float x = pb[n], y = pb[NN + n], z = pb[2 * NN + n];
  sq[pid] = x * x + y * y + z * z;
}

// ---------------- 2) brute-force kNN (K=16 smallest) ----------------
__global__ void knn_kernel(const float* __restrict__ pos, const float* __restrict__ sq,
                           int* __restrict__ idxOut) {
  int pid = blockIdx.x * blockDim.x + threadIdx.x;
  if (pid >= BN) return;
  int b = pid / NN, i = pid % NN;
  const float* pb = pos + (size_t)b * 3 * NN;
  const float* sb = sq + (size_t)b * NN;
  float qx = pb[i], qy = pb[NN + i], qz = pb[2 * NN + i];
  float sqi = sb[i];

  float bd[KK];
  int   bi[KK];
#pragma unroll
  for (int k = 0; k < KK; ++k) { bd[k] = 3.4e38f; bi[k] = 0; }

  for (int j = 0; j < NN; ++j) {
    float d = sqi + sb[j] - 2.f * (qx * pb[j] + qy * pb[NN + j] + qz * pb[2 * NN + j]);
    if (d < bd[KK - 1]) {
      int p = KK - 1;
      while (p > 0 && bd[p - 1] > d) { bd[p] = bd[p - 1]; bi[p] = bi[p - 1]; --p; }
      bd[p] = d; bi[p] = j;
    }
  }
#pragma unroll
  for (int k = 0; k < KK; ++k) idxOut[(size_t)pid * KK + k] = bi[k];
}

// ---------------- 3) BN statistics (partial, deterministic) ----------------
// 11 accumulators: [sx, sxx, s0, s1, s2, p00, p01, p02, p11, p12, p22]
__global__ void stats_partial_kernel(const float* __restrict__ pos,
                                     const float* __restrict__ invd,
                                     const int* __restrict__ nn_idx,
                                     float* __restrict__ part) {
  int t = threadIdx.x;
  int pid = blockIdx.x * 256 + t;           // grid = BN/256 = 64 blocks
  int b = pid / NN, n = pid % NN;
  const float* pb = pos + (size_t)b * 3 * NN;
  const float* ib = invd + (size_t)b * NN;
  const int* idx = nn_idx + (size_t)pid * KK;

  int id[KK];
  float mx = 0.f;
#pragma unroll
  for (int k = 0; k < KK; ++k) { id[k] = idx[k]; mx = fmaxf(mx, ib[id[k]]); }

  float sx = 0.f, sxx = 0.f;
  float s0 = 0.f, s1 = 0.f, s2 = 0.f;
  float p00 = 0.f, p01 = 0.f, p02 = 0.f, p11 = 0.f, p12 = 0.f, p22 = 0.f;
  float cx = pb[n], cy = pb[NN + n], cz = pb[2 * NN + n];
#pragma unroll
  for (int k = 0; k < KK; ++k) {
    int j = id[k];
    float x = ib[j] / mx;
    sx += x; sxx += x * x;
    float d0 = pb[j] - cx, d1 = pb[NN + j] - cy, d2 = pb[2 * NN + j] - cz;
    s0 += d0; s1 += d1; s2 += d2;
    p00 += d0 * d0; p01 += d0 * d1; p02 += d0 * d2;
    p11 += d1 * d1; p12 += d1 * d2; p22 += d2 * d2;
  }
  float vals[11] = {sx, sxx, s0, s1, s2, p00, p01, p02, p11, p12, p22};

  __shared__ float red[256];
  for (int i = 0; i < 11; ++i) {
    red[t] = vals[i];
    __syncthreads();
    for (int s = 128; s > 0; s >>= 1) {
      if (t < s) red[t] += red[t + s];
      __syncthreads();
    }
    if (t == 0) part[blockIdx.x * 11 + i] = red[0];
    __syncthreads();
  }
}

// ---------------- 4) fold BN analytically into per-channel affine ----------------
// coef layout: dnA[16] | dnB[16] | wnA[32*3] | wnB[32]   (160 floats)
__global__ void stats_coeffs_kernel(const float* __restrict__ part,
                                    const float* __restrict__ dn_w1, const float* __restrict__ dn_g1,
                                    const float* __restrict__ dn_be1,
                                    const float* __restrict__ wn_w,  const float* __restrict__ wn_g,
                                    const float* __restrict__ wn_be,
                                    float* __restrict__ coef) {
  __shared__ float tot[11];
  int t = threadIdx.x;
  if (t < 11) {
    float s = 0.f;
    for (int blk = 0; blk < BN / 256; ++blk) s += part[blk * 11 + t];
    tot[t] = s;
  }
  __syncthreads();
  const float cnt = (float)(BB * NN * KK);
  if (t < DN) {
    float mxv = tot[0] / cnt;
    float vx = tot[1] / cnt - mxv * mxv;
    float w = dn_w1[t];
    float rs = rsqrtf(w * w * vx + EPSF);
    float a = dn_g1[t] * w * rs;
    coef[t] = a;                                 // dnA
    coef[DN + t] = dn_be1[t] - a * mxv;          // dnB (conv bias cancels in BN)
  }
  if (t < WW) {
    float m0 = tot[2] / cnt, m1 = tot[3] / cnt, m2 = tot[4] / cnt;
    float C00 = tot[5] / cnt - m0 * m0, C01 = tot[6] / cnt - m0 * m1, C02 = tot[7] / cnt - m0 * m2;
    float C11 = tot[8] / cnt - m1 * m1, C12 = tot[9] / cnt - m1 * m2, C22 = tot[10] / cnt - m2 * m2;
    float w0 = wn_w[t * 3 + 0], w1 = wn_w[t * 3 + 1], w2 = wn_w[t * 3 + 2];
    float vy = w0 * w0 * C00 + w1 * w1 * C11 + w2 * w2 * C22
             + 2.f * (w0 * w1 * C01 + w0 * w2 * C02 + w1 * w2 * C12);
    float rs = rsqrtf(vy + EPSF);
    float g = wn_g[t];
    float A0 = g * w0 * rs, A1 = g * w1 * rs, A2 = g * w2 * rs;
    coef[2 * DN + t * 3 + 0] = A0;
    coef[2 * DN + t * 3 + 1] = A1;
    coef[2 * DN + t * 3 + 2] = A2;
    coef[2 * DN + 3 * WW + t] = wn_be[t] - (A0 * m0 + A1 * m1 + A2 * m2);
  }
}

// ---------------- 4b) pre-convert + permute lin_w to f16 -------------------
// linP[co][cc][w][cl] = lin_w[co][(cc*16+cl)*32 + w]  (f16)
// so GEMM2's K ordering kk' = w*16+cl matches featT's [p][w][cl] layout.
__global__ void linperm_kernel(const float* __restrict__ lin_w, _Float16* __restrict__ linP) {
  int i = blockIdx.x * 256 + threadIdx.x;
  if (i >= LINSZ) return;
  int co  = i >> 11;
  int rem = i & 2047;
  int ccv = rem >> 9;
  int w   = (rem >> 4) & 31;
  int cl  = rem & 15;
  int ksrc = (ccv * 16 + cl) * 32 + w;
  linP[i] = (_Float16)lin_w[(size_t)co * 2048 + ksrc];
}

// ---------------- 5) fused density/weight/matmul/linear per 16-point tile ----
__global__ __launch_bounds__(128)
void fused_kernel(const float* __restrict__ inputs,   // (B,C,N)
                  const float* __restrict__ pos,      // (B,3,N)
                  const float* __restrict__ invd,     // (B,1,N)
                  const int*   __restrict__ nn_idx,   // (B*N,K)
                  const float* __restrict__ coef,     // folded BN coeffs
                  const float* __restrict__ dn_w2,    // (1,16)
                  const float* __restrict__ dn_b2,    // (1,)
                  const _Float16* __restrict__ linP,  // permuted f16 lin_w
                  const float* __restrict__ lin_b,    // (COUT,)
                  float* __restrict__ out) {          // (B,COUT,N)
  __shared__ int      idxS[TP * KK];                       // 1 KB
  __shared__ float    maxinvS[TP];
  __shared__ float    sdS[TP * KK];                        // 1 KB
  __shared__ float    lnnS[TP * KK * 3];                   // 3 KB
  __shared__ __align__(16) _Float16 zeroS[16];             // zero K>=16 source
  __shared__ __align__(16) _Float16 WtT[TP * WW * KK];     // [p][o][k], 16 KB
  __shared__ __align__(16) _Float16 SchS[TP * 16 * KK];    // [p][cl][k], 8 KB
  __shared__ __align__(16) _Float16 featT[TP * WW * 16];   // [p][w][cl], 16 KB

  const int t = threadIdx.x;
  const int wave = t >> 5;
  const int lane = t & 31;
  const int m_lane = lane & 15;
  const int half = lane >> 4;

  const int bId = blockIdx.x;                 // B * (N/TP) = 1024
  const int b = bId / (NN / TP);
  const int n0 = (bId % (NN / TP)) * TP;

  const float* inpB  = inputs + (size_t)b * CC * NN;
  const float* posB  = pos + (size_t)b * 3 * NN;
  const float* invdB = invd + (size_t)b * NN;
  const float* dnA = coef;
  const float* dnB = coef + DN;
  const float* wnA = coef + 2 * DN;
  const float* wnB = coef + 2 * DN + 3 * WW;

  // ---- Phase A: indices, density scale, translated positions, WeightNet ----
  if (t < 16) zeroS[t] = (_Float16)0.f;
  for (int v = t; v < TP * KK; v += 128)
    idxS[v] = nn_idx[((size_t)(b * NN + n0 + (v >> 4))) * KK + (v & 15)];
  __syncthreads();

  if (t < TP) {
    float mx = 0.f;
#pragma unroll
    for (int k = 0; k < KK; ++k) mx = fmaxf(mx, invdB[idxS[t * KK + k]]);
    maxinvS[t] = mx;
  }
  __syncthreads();

  for (int v = t; v < TP * KK; v += 128) {
    int p = v >> 4;
    int j = idxS[v];
    float x = invdB[j] / maxinvS[p];
    float acc = dn_b2[0];
#pragma unroll
    for (int c = 0; c < DN; ++c)
      acc += dn_w2[c] * fmaxf(0.f, dnA[c] * x + dnB[c]);
    sdS[v] = 1.f / (1.f + __expf(-acc));
    int n = n0 + p;
    lnnS[v * 3 + 0] = posB[j] - posB[n];
    lnnS[v * 3 + 1] = posB[NN + j] - posB[NN + n];
    lnnS[v * 3 + 2] = posB[2 * NN + j] - posB[2 * NN + n];
  }
  __syncthreads();

  for (int v = t; v < TP * KK * WW; v += 128) {
    int k = v & 15;
    int o = (v >> 4) & 31;
    int p = v >> 9;
    int pk = p * 16 + k;
    float y = wnB[o] + wnA[o * 3 + 0] * lnnS[pk * 3 + 0]
                     + wnA[o * 3 + 1] * lnnS[pk * 3 + 1]
                     + wnA[o * 3 + 2] * lnnS[pk * 3 + 2];
    WtT[p * 512 + o * 16 + k] = (_Float16)fmaxf(0.f, y);   // [p][o][k]
  }
  __syncthreads();

  // ---- Phases B/C: chunked GEMM1 (per-point 16x16x32) feeding GEMM2 -------
  v8f acc = {};                         // this wave's 16-point x 16-cout tile
  const int co = wave * 16 + m_lane;    // GEMM2 output column
  for (int cc = 0; cc < CC / 16; ++cc) {
    // gather & scale input chunk: S[p][cl][k] = inputs[b, cc*16+cl, idx] * sd
    for (int v = t; v < TP * 16 * KK; v += 128) {
      int k = v & 15, cl = (v >> 4) & 15, p = v >> 8;
      int c = cc * 16 + cl;
      float val = inpB[(size_t)c * NN + idxS[p * KK + k]] * sdS[p * KK + k];
      SchS[p * 256 + cl * 16 + k] = (_Float16)val;
    }
    __syncthreads();

    // GEMM1: each wave handles 4 points; (16c x 16k) @ (16k x 32w), K padded
    for (int pi = 0; pi < 4; ++pi) {
      int p = wave * 4 + pi;
      // A: one contiguous b128 per lane; upper 8 elements (K>=16) are zero
      v8h alo = *(const v8h*)(SchS + p * 256 + m_lane * 16 + half * 8);
      v16h a = {};
#pragma unroll
      for (int e = 0; e < 8; ++e) a[e] = alo[e];
#pragma unroll
      for (int nt = 0; nt < 2; ++nt) {
        // B: lower 8 elements k = 8*half + e; upper 8 (K>=16) read zeroed LDS
        const _Float16* bsrc = WtT + p * 512 + (nt * 16 + m_lane) * 16 + half * 8;
        v8h blo = *(const v8h*)bsrc;
        v8h bhi = *(const v8h*)zeroS;
        v16h bv;
#pragma unroll
        for (int e = 0; e < 8; ++e) { bv[e] = blo[e]; bv[e + 8] = bhi[e]; }
        v8f z = {};
        v8f d = wmma_f16(a, bv, z);
        // D rows cl = r + 8*half -> contiguous in featT[p][w][cl]: 1x b128
        v8h hv;
#pragma unroll
        for (int r = 0; r < 8; ++r) hv[r] = (_Float16)d[r];
        *(v8h*)(featT + p * 512 + (nt * 16 + m_lane) * 16 + 8 * half) = hv;
      }
    }
    __syncthreads();

    // GEMM2 partial: (16 points x 512) @ (512 x 16 couts), wave owns cout tile
    if (cc < CC / 16 - 1)
      __builtin_prefetch(linP + (size_t)co * 2048 + (cc + 1) * 512, 0, 1);
    for (int ks = 0; ks < 16; ++ks) {
      const _Float16* abase = featT + m_lane * 512 + ks * 32 + half * 8;
      v8h alo = *(const v8h*)abase;
      v8h ahi = *(const v8h*)(abase + 16);
      const _Float16* bbase = linP + (size_t)co * 2048 + cc * 512 + ks * 32 + half * 8;
      v8h blo = *(const v8h*)bbase;
      v8h bhi = *(const v8h*)(bbase + 16);
      v16h a, bv;
#pragma unroll
      for (int e = 0; e < 8; ++e) {
        a[e] = alo[e];  a[e + 8]  = ahi[e];
        bv[e] = blo[e]; bv[e + 8] = bhi[e];
      }
      acc = wmma_f16(a, bv, acc);
    }
    __syncthreads();
  }

  // ---- write out (B, COUT, N): 8 consecutive n per lane -> two b128 -------
  const float bias = lin_b[co];
  float* obase = out + (size_t)b * COUT * NN + (size_t)co * NN + n0 + 8 * half;
  float4 o0 = {acc[0] + bias, acc[1] + bias, acc[2] + bias, acc[3] + bias};
  float4 o1 = {acc[4] + bias, acc[5] + bias, acc[6] + bias, acc[7] + bias};
  *(float4*)(obase)     = o0;
  *(float4*)(obase + 4) = o1;
}

// ---------------- launcher ----------------
extern "C" void kernel_launch(void* const* d_in, const int* in_sizes, int n_in,
                              void* d_out, int out_size, void* d_ws, size_t ws_size,
                              hipStream_t stream) {
  const float* inputs = (const float*)d_in[0];
  const float* pos    = (const float*)d_in[1];
  const float* invd   = (const float*)d_in[2];
  const float* wn_w   = (const float*)d_in[3];
  const float* wn_g   = (const float*)d_in[5];
  const float* wn_be  = (const float*)d_in[6];
  const float* dn_w1  = (const float*)d_in[7];
  const float* dn_g1  = (const float*)d_in[9];
  const float* dn_be1 = (const float*)d_in[10];
  const float* dn_w2  = (const float*)d_in[11];
  const float* dn_b2  = (const float*)d_in[12];
  const float* lin_w  = (const float*)d_in[13];
  const float* lin_b  = (const float*)d_in[14];
  float* out = (float*)d_out;

  // workspace layout
  float* ws_sq   = (float*)d_ws;                         // BN floats
  int*   ws_idx  = (int*)(ws_sq + BN);                   // BN*K ints
  float* ws_part = (float*)(ws_idx + (size_t)BN * KK);   // 64*11 floats
  float* ws_coef = ws_part + (BN / 256) * 11;            // 160 floats
  _Float16* ws_linP = (_Float16*)(ws_coef + 160);        // 64*2048 f16

  sqnorm_kernel<<<BN / 256, 256, 0, stream>>>(pos, ws_sq);
  knn_kernel<<<BN / 256, 256, 0, stream>>>(pos, ws_sq, ws_idx);
  stats_partial_kernel<<<BN / 256, 256, 0, stream>>>(pos, invd, ws_idx, ws_part);
  stats_coeffs_kernel<<<1, 128, 0, stream>>>(ws_part, dn_w1, dn_g1, dn_be1,
                                             wn_w, wn_g, wn_be, ws_coef);
  linperm_kernel<<<(LINSZ + 255) / 256, 256, 0, stream>>>(lin_w, ws_linP);
  fused_kernel<<<BB * (NN / TP), 128, 0, stream>>>(inputs, pos, invd, ws_idx, ws_coef,
                                                   dn_w2, dn_b2, ws_linP, lin_b, out);
}
